// CDAN_Dis_18373870092946
// MI455X (gfx1250) — matrix operand order repaired and checked by reading, not compiled, not run on hardware
//
#include <hip/hip_runtime.h>
#include <hip/hip_bf16.h>
#include <stddef.h>

// ---------------------------------------------------------------------------
// Problem constants (from reference): M(batch)=4, C=2, B(ch)=128, T=2000
// conv1: 256ch, T1=1000, K1=384 ; conv2: 256ch, T2=500, K2=768 ; out 4x1x500
// ---------------------------------------------------------------------------
#define NB   4
#define CB   128
#define TT   2000
#define C1   256
#define T1   1000
#define K1   384
#define KC1  12      // K1/32
#define NT1  63      // ceil(T1/16)
#define C2   256
#define T2   500
#define K2   768
#define KC2  24      // K2/32
#define NT2  32      // ceil(T2/16)
#define EPSV 1e-8f

typedef _Float16 v16h __attribute__((ext_vector_type(16)));
typedef float    v8f  __attribute__((ext_vector_type(8)));

__device__ __forceinline__ float leaky_f(float x) { return x >= 0.f ? x : 0.1f * x; }

// ---------------------------------------------------------------------------
// 1) a[b,c,t] = sum_f w2d[c,f] * feature[b,f,t]
// ---------------------------------------------------------------------------
__global__ void cd_mix_a(const float* __restrict__ feature,
                         const float* __restrict__ w2d,
                         float* __restrict__ a_buf) {
    int id = blockIdx.x * blockDim.x + threadIdx.x;
    if (id >= NB * 2 * TT) return;
    int t = id % TT;
    int c = (id / TT) & 1;
    int b = id / (2 * TT);
    float s = 0.f;
    #pragma unroll 4
    for (int f = 0; f < CB; ++f)
        s += w2d[c * CB + f] * feature[((size_t)(b * CB + f)) * TT + t];
    a_buf[id] = s;   // layout (b*2 + c)*TT + t
}

// ---------------------------------------------------------------------------
// 2) d[b,p,t] = a0*mask0 + a1*mask1 + b2d ; block-partial sum/sumsq per batch
//    grid = (16, NB), block = 256
// ---------------------------------------------------------------------------
__global__ __launch_bounds__(256)
void cd_mix_d(const float* __restrict__ a_buf, const float* __restrict__ mask,
              const float* __restrict__ b2d, float* __restrict__ d,
              float* __restrict__ partials) {
    const int b   = blockIdx.y;
    const int tid = threadIdx.x;
    const float* ab = a_buf + b * 2 * TT;
    const float  bb = b2d[0];
    float s = 0.f, s2 = 0.f;
    const int N = CB * TT;                    // 256000 per batch
    for (int g = blockIdx.x * 256 + tid; g < N; g += 16 * 256) {
        int p = g / TT, t = g % TT;
        float dv = ab[t]      * mask[((size_t)((b * 2 + 0) * CB + p)) * TT + t]
                 + ab[TT + t] * mask[((size_t)((b * 2 + 1) * CB + p)) * TT + t]
                 + bb;
        d[(size_t)b * N + g] = dv;
        s += dv; s2 += dv * dv;
    }
    __shared__ float ss[256], sq[256];
    ss[tid] = s; sq[tid] = s2; __syncthreads();
    for (int o = 128; o > 0; o >>= 1) {
        if (tid < o) { ss[tid] += ss[tid + o]; sq[tid] += sq[tid + o]; }
        __syncthreads();
    }
    if (tid == 0) {
        int bi = b * 16 + blockIdx.x;
        partials[bi * 2] = ss[0]; partials[bi * 2 + 1] = sq[0];
    }
}

// ---------------------------------------------------------------------------
// 3) deterministic stats: mean & rstd per batch from fixed-order partials
// ---------------------------------------------------------------------------
__global__ void cd_reduce_stats(const float* __restrict__ partials, int nper,
                                float invN, float* __restrict__ stats_out) {
    int b = threadIdx.x;
    if (b >= NB) return;
    float s = 0.f, s2 = 0.f;
    for (int p = 0; p < nper; ++p) {
        s  += partials[(b * nper + p) * 2];
        s2 += partials[(b * nper + p) * 2 + 1];
    }
    float mean = s * invN;
    float var  = s2 * invN - mean * mean;
    stats_out[b * 2]     = mean;
    stats_out[b * 2 + 1] = 1.f / sqrtf(var + EPSV);
}

// ---------------------------------------------------------------------------
// 4) pack conv weights (Cout x Cin x 3) into WMMA 16-bit A-matrix lane layout.
//    Storage: Ap[(((mt*KC)+kc)*32 + lane)*16 + e], e holds K = 16*h + e of the 16x32 A tile.
// ---------------------------------------------------------------------------
__global__ void cd_pack_w(const float* __restrict__ w, _Float16* __restrict__ Ap,
                          int KC, int Cin) {
    int total = 16 * KC * 32 * 16;
    int id = blockIdx.x * blockDim.x + threadIdx.x;
    if (id >= total) return;
    int e    = id & 15;
    int lane = (id >> 4) & 31;
    int rest = id >> 9;
    int kc   = rest % KC;
    int mt   = rest / KC;
    int m = lane & 15, h = lane >> 4;
    // contiguous K order (matches B operand): element e of lane-half h holds K = 16h+e
    int K = kc * 32 + 16 * h + e;
    int o = mt * 16 + m;
    int i = K / 3, k = K % 3;
    Ap[id] = (_Float16)w[((size_t)o * Cin + i) * 3 + k];
}

// ---------------------------------------------------------------------------
// 5) pack im2col activations into WMMA 16-bit B-matrix rows (t-major):
//    X[(b*Tout + t)*Kdim + kappa] = f16( leaky( LN(x[b, kappa/3, 2t-1+kappa%3]) ) )
//    scalar-gamma variant (stage0 -> conv1 input)
// ---------------------------------------------------------------------------
__global__ void cd_pack_x1(const float* __restrict__ d, const float* __restrict__ stats,
                           const float* __restrict__ g, const float* __restrict__ be,
                           _Float16* __restrict__ X) {
    int total = NB * T1 * K1;
    int id = blockIdx.x * blockDim.x + threadIdx.x;
    if (id >= total) return;
    int kap = id % K1;
    int t   = (id / K1) % T1;
    int b   = id / (K1 * T1);
    int i = kap / 3, k = kap % 3;
    int tin = 2 * t - 1 + k;
    float v = 0.f;
    if (tin >= 0 && tin < TT) {
        float mean = stats[b * 2], rstd = stats[b * 2 + 1];
        float dv = d[((size_t)(b * CB + i)) * TT + tin];
        v = leaky_f(g[0] * (dv - mean) * rstd + be[0]);
    }
    X[id] = (_Float16)v;
}

// per-channel-gamma variant (stage1 -> conv2 input)
__global__ void cd_pack_x2(const float* __restrict__ y1, const float* __restrict__ stats,
                           const float* __restrict__ g, const float* __restrict__ be,
                           _Float16* __restrict__ X) {
    int total = NB * T2 * K2;
    int id = blockIdx.x * blockDim.x + threadIdx.x;
    if (id >= total) return;
    int kap = id % K2;
    int t   = (id / K2) % T2;
    int b   = id / (K2 * T2);
    int i = kap / 3, k = kap % 3;
    int tin = 2 * t - 1 + k;
    float v = 0.f;
    if (tin >= 0 && tin < T1) {
        float mean = stats[b * 2], rstd = stats[b * 2 + 1];
        float yv = y1[((size_t)(b * C1 + i)) * T1 + tin];
        v = leaky_f(g[i] * (yv - mean) * rstd + be[i]);
    }
    X[id] = (_Float16)v;
}

// ---------------------------------------------------------------------------
// 6) WMMA conv GEMM: one wave = one 16(M)x16(N) f32 tile, KC chained
//    v_wmma_f32_16x16x32_f16 with f32 accumulation; fused bias + LN partials.
//    grid = (Ntiles, 2, NB), block = 256 (8 waves; wave w -> mtile blockIdx.y*8+w)
// ---------------------------------------------------------------------------
template <int KC, int TOUT>
__global__ __launch_bounds__(256)
void cd_conv_wmma(const _Float16* __restrict__ Ap, const _Float16* __restrict__ Xp,
                  const float* __restrict__ bias, float* __restrict__ Y,
                  float* __restrict__ partials) {
    const int tid  = threadIdx.x;
    const int wv   = tid >> 5;
    const int lane = tid & 31;
    const int n    = lane & 15;    // N column within tile
    const int h    = lane >> 4;    // lane-half
    const int b    = blockIdx.z;
    const int mt   = blockIdx.y * 8 + wv;
    const int t0   = blockIdx.x * 16;
    const int Kdim = KC * 32;

    const int t   = t0 + n;
    const int tcl = (t < TOUT) ? t : (TOUT - 1);   // clamp tail loads (excluded below)

    const v16h* ap = (const v16h*)Ap + ((size_t)mt * KC) * 32 + lane;
    const v16h* xp = (const v16h*)Xp + ((size_t)(b * TOUT + tcl) * Kdim) / 16 + h;

    v8f acc = {};
    #pragma unroll
    for (int kc = 0; kc < KC; ++kc) {
        v16h av = ap[(size_t)kc * 32];   // A 16x32 tile, this lane's 16 halves
        v16h bv = xp[kc * 2];            // B 32x16 tile: K = 32*kc + 16*h + e, col n
        acc = __builtin_amdgcn_wmma_f32_16x16x32_f16(
                false, av, false, bv, (short)0, acc, false, false);
    }

    float s = 0.f, s2 = 0.f;
    const bool valid = (t < TOUT);
    #pragma unroll
    for (int r = 0; r < 8; ++r) {
        int o = mt * 16 + r + 8 * h;     // D row layout: VGPR r -> M = r + 8*half
        float y = acc[r] + bias[o];
        if (valid) {
            Y[((size_t)(b * C1 + o)) * TOUT + t] = y;
            s += y; s2 += y * y;
        }
    }
    __shared__ float ss[256], sq[256];
    ss[tid] = s; sq[tid] = s2; __syncthreads();
    for (int o = 128; o > 0; o >>= 1) {
        if (tid < o) { ss[tid] += ss[tid + o]; sq[tid] += sq[tid + o]; }
        __syncthreads();
    }
    if (tid == 0) {
        int blkPerB = gridDim.x * gridDim.y;
        int bi = b * blkPerB + blockIdx.y * gridDim.x + blockIdx.x;
        partials[bi * 2] = ss[0]; partials[bi * 2 + 1] = sq[0];
    }
}

// ---------------------------------------------------------------------------
// 7) final 1x1 conv: out[b,t] = b3 + sum_i w3[i] * leaky(LN2(y2[b,i,t]))
//    one block per (b,t), 256-thread tree reduction (deterministic)
// ---------------------------------------------------------------------------
__global__ __launch_bounds__(256)
void cd_final(const float* __restrict__ y2, const float* __restrict__ stats,
              const float* __restrict__ g, const float* __restrict__ be,
              const float* __restrict__ w3, const float* __restrict__ b3,
              float* __restrict__ out) {
    int bt = blockIdx.x;
    int b = bt / T2, t = bt % T2;
    int i = threadIdx.x;
    float mean = stats[b * 2], rstd = stats[b * 2 + 1];
    float v  = y2[((size_t)(b * C2 + i)) * T2 + t];
    float xn = leaky_f(g[i] * (v - mean) * rstd + be[i]);
    __shared__ float ss[256];
    ss[i] = w3[i] * xn; __syncthreads();
    for (int o = 128; o > 0; o >>= 1) {
        if (i < o) ss[i] += ss[i + o];
        __syncthreads();
    }
    if (i == 0) out[bt] = ss[0] + b3[0];
}

// ---------------------------------------------------------------------------
extern "C" void kernel_launch(void* const* d_in, const int* in_sizes, int n_in,
                              void* d_out, int out_size, void* d_ws, size_t ws_size,
                              hipStream_t stream) {
    const float* feature = (const float*)d_in[0];
    const float* mask    = (const float*)d_in[1];
    const float* w2d     = (const float*)d_in[2];
    const float* b2d     = (const float*)d_in[3];
    const float* g2d     = (const float*)d_in[4];
    const float* be2d    = (const float*)d_in[5];
    const float* w1      = (const float*)d_in[6];
    const float* b1      = (const float*)d_in[7];
    const float* g1      = (const float*)d_in[8];
    const float* bb1     = (const float*)d_in[9];
    const float* w2      = (const float*)d_in[10];
    const float* b2      = (const float*)d_in[11];
    const float* g2      = (const float*)d_in[12];
    const float* bb2     = (const float*)d_in[13];
    const float* w3      = (const float*)d_in[14];
    const float* b3      = (const float*)d_in[15];
    float* out = (float*)d_out;

    // --- carve workspace (256B-aligned regions), total ~17 MB ---
    char* wp = (char*)d_ws;
    auto carve = [&](size_t bytes) -> void* {
        void* p = (void*)wp;
        wp += (bytes + 255) & ~(size_t)255;
        return p;
    };
    float*    a_buf = (float*)carve((size_t)NB * 2 * TT * 4);
    float*    st0   = (float*)carve(8 * 4);
    float*    st1   = (float*)carve(8 * 4);
    float*    st2   = (float*)carve(8 * 4);
    float*    part0 = (float*)carve((size_t)NB * 16  * 2 * 4);
    float*    part1 = (float*)carve((size_t)NB * (NT1 * 2) * 2 * 4);
    float*    part2 = (float*)carve((size_t)NB * (NT2 * 2) * 2 * 4);
    float*    dbuf  = (float*)carve((size_t)NB * CB * TT * 4);
    float*    y1    = (float*)carve((size_t)NB * C1 * T1 * 4);
    float*    y2    = (float*)carve((size_t)NB * C2 * T2 * 4);
    _Float16* Ap1   = (_Float16*)carve((size_t)16 * KC1 * 32 * 16 * 2);
    _Float16* Ap2   = (_Float16*)carve((size_t)16 * KC2 * 32 * 16 * 2);
    _Float16* X1    = (_Float16*)carve((size_t)NB * T1 * K1 * 2);
    _Float16* X2    = (_Float16*)carve((size_t)NB * T2 * K2 * 2);

    // --- stage 0: mixing + global LN stats ---
    cd_mix_a<<<(NB * 2 * TT + 255) / 256, 256, 0, stream>>>(feature, w2d, a_buf);
    cd_mix_d<<<dim3(16, NB), 256, 0, stream>>>(a_buf, mask, b2d, dbuf, part0);
    cd_reduce_stats<<<1, NB, 0, stream>>>(part0, 16, 1.f / (CB * TT), st0);

    // --- stage 1: conv1 as WMMA GEMM (M=256, N=1000, K=384) ---
    cd_pack_w<<<(16 * KC1 * 32 * 16 + 255) / 256, 256, 0, stream>>>(w1, Ap1, KC1, CB);
    cd_pack_x1<<<(NB * T1 * K1 + 255) / 256, 256, 0, stream>>>(dbuf, st0, g2d, be2d, X1);
    cd_conv_wmma<KC1, T1><<<dim3(NT1, 2, NB), 256, 0, stream>>>(Ap1, X1, b1, y1, part1);
    cd_reduce_stats<<<1, NB, 0, stream>>>(part1, NT1 * 2, 1.f / (C1 * T1), st1);

    // --- stage 2: conv2 as WMMA GEMM (M=256, N=500, K=768) ---
    cd_pack_w<<<(16 * KC2 * 32 * 16 + 255) / 256, 256, 0, stream>>>(w2, Ap2, KC2, C1);
    cd_pack_x2<<<(NB * T2 * K2 + 255) / 256, 256, 0, stream>>>(y1, st1, g1, bb1, X2);
    cd_conv_wmma<KC2, T2><<<dim3(NT2, 2, NB), 256, 0, stream>>>(Ap2, X2, b2, y2, part2);
    cd_reduce_stats<<<1, NB, 0, stream>>>(part2, NT2 * 2, 1.f / (C2 * T2), st2);

    // --- stage 3: 1x1 conv head ---
    cd_final<<<NB * T2, 256, 0, stream>>>(y2, st2, g2, bb2, w3, b3, out);
    (void)in_sizes; (void)n_in; (void)out_size; (void)ws_size;
}
